// LSTMRegression_39908836114819
// MI455X (gfx1250) — matrix operand order repaired and checked by reading, not compiled, not run on hardware
//
#include <hip/hip_runtime.h>
#include <hip/hip_bf16.h>

typedef __attribute__((ext_vector_type(16))) _Float16 v16h;
typedef __attribute__((ext_vector_type(8)))  _Float16 v8h;
typedef __attribute__((ext_vector_type(8)))  float    v8f;

union FragA { v16h v; v8h h[2]; };

__device__ __forceinline__ v8f vzero8() {
    v8f z = {0.f, 0.f, 0.f, 0.f, 0.f, 0.f, 0.f, 0.f};
    return z;
}
__device__ __forceinline__ float sigm(float x) { return 1.f / (1.f + __expf(-x)); }
__device__ __forceinline__ float tanh_fast(float x) { return 2.f / (1.f + __expf(-2.f * x)) - 1.f; }

// ---------------------------------------------------------------------------
// fp32 -> fp16 elementwise convert
__global__ void k_cvt_f16(const float* __restrict__ src, _Float16* __restrict__ dst, size_t n) {
    size_t i = (size_t)blockIdx.x * blockDim.x + threadIdx.x;
    if (i < n) dst[i] = (_Float16)src[i];
}

// ---------------------------------------------------------------------------
// Pack fp32 weight W[N][K] (row-major) into f16 WMMA B-fragment order.
// Fragment f = n16*(K/32)+kk holds B[k][n] for k in [kk*32,+32), n in [n16*16,+16).
// B layout (16-bit, 32x16): lane%16 = local n, elem e -> local k = 8*(lane/16) + (e&7) + 16*(e>>3).
__global__ void k_pack_w(const float* __restrict__ W, _Float16* __restrict__ out, int N, int K) {
    size_t idx = (size_t)blockIdx.x * blockDim.x + threadIdx.x;
    size_t total = (size_t)N * K;
    if (idx >= total) return;
    int KK = K >> 5;
    size_t f = idx >> 9;
    int r = (int)(idx & 511);
    int lane = r >> 4;
    int e = r & 15;
    int n16 = (int)(f / KK);
    int kk = (int)(f % KK);
    out[idx] = (_Float16)W[(size_t)(n16 * 16 + (lane & 15)) * K + (size_t)kk * 32 + 8 * (lane >> 4) + (e & 7) + 16 * (e >> 3)];
}

// ---------------------------------------------------------------------------
__global__ void k_zero_u32(unsigned int* __restrict__ p, size_t n) {
    size_t i = (size_t)blockIdx.x * blockDim.x + threadIdx.x;
    if (i < n) p[i] = 0u;
}

// ---------------------------------------------------------------------------
// Big GEMM, 64x64 per wave (4 m-tiles x 4 n-tiles, 16 accumulators).
// out[t][b][n] = (f16)( X[m][:] . W^T[:, n] + bias[n] )
// M = 32768 (=B*T), N = 4096, K in {256,1024}.
// rows_bt==1: row m -> (b=m/512, t=m%512); rows_bt==0: row m -> (t=m/64, b=m%64).
__global__ __launch_bounds__(256) void k_gemm_xg(
    const _Float16* __restrict__ X, const _Float16* __restrict__ Wp,
    const float* __restrict__ bias, _Float16* __restrict__ out,
    int K, int rows_bt)
{
    const int lane = threadIdx.x & 31;
    const int wave = (int)((blockIdx.x * blockDim.x + threadIdx.x) >> 5); // 0..32767
    const int n64 = wave & 63;       // 64-col group
    const int mg  = wave >> 6;       // 0..511 -> 64-row group
    const int m0 = mg * 64;
    const int n0 = n64 * 64;
    const int lo = lane >> 4;
    const int nl = lane & 15;
    const int l15 = lane & 15;
    const int KK = K >> 5;

    v8f acc[4][4];
    #pragma unroll
    for (int a = 0; a < 4; ++a)
        #pragma unroll
        for (int j = 0; j < 4; ++j) acc[a][j] = vzero8();

    // base pointers; inner loop uses pure immediate offsets after strength reduction
    const _Float16* ap[4];
    #pragma unroll
    for (int a = 0; a < 4; ++a)
        ap[a] = X + (size_t)(m0 + 16 * a + l15) * K + 8 * lo;
    const _Float16* bp[4];
    #pragma unroll
    for (int j = 0; j < 4; ++j)
        bp[j] = Wp + ((size_t)((n64 * 4 + j) * KK) << 9) + lane * 16;

    #pragma unroll 2
    for (int kk = 0; kk < KK; ++kk) {
        FragA A[4], Bv[4];
        #pragma unroll
        for (int a = 0; a < 4; ++a) {
            A[a].h[0] = *(const v8h*)(ap[a] + kk * 32);
            A[a].h[1] = *(const v8h*)(ap[a] + kk * 32 + 16);
        }
        #pragma unroll
        for (int j = 0; j < 4; ++j) {
            Bv[j].h[0] = *(const v8h*)(bp[j] + kk * 512);
            Bv[j].h[1] = *(const v8h*)(bp[j] + kk * 512 + 8);
        }
        #pragma unroll
        for (int a = 0; a < 4; ++a)
            #pragma unroll
            for (int j = 0; j < 4; ++j)
                acc[a][j] = __builtin_amdgcn_wmma_f32_16x16x32_f16(
                    false, A[a].v, false, Bv[j].v, (short)0, acc[a][j], false, false);
    }

    // epilogue: lane -> N = n0+16j+nl, elem e -> M = m0+16a+8*lo+e
    #pragma unroll
    for (int j = 0; j < 4; ++j) {
        const int n = n0 + j * 16 + nl;
        const float bv = bias[n];
        #pragma unroll
        for (int a = 0; a < 4; ++a) {
            #pragma unroll
            for (int e = 0; e < 8; ++e) {
                const int m = m0 + 16 * a + 8 * lo + e;
                int t, b;
                if (rows_bt) { b = m >> 9; t = m & 511; }
                else         { t = m >> 6; b = m & 63;  }
                out[((size_t)t * 64 + b) * 4096 + n] = (_Float16)(acc[a][j][e] + bv);
            }
        }
    }
}

// ---------------------------------------------------------------------------
// Persistent recurrence: 64 WGs x 128 threads (4 waves).
// WG w owns h-columns [16w,16w+16); wave v owns batch rows [16v,16v+16).
// The WG's Whh slice (4 gates x 16 cols x K=1024 f16 = 128KB) is staged in LDS
// ONCE, then all 512 steps read B fragments via ds_load_b128 (imm offsets).
__global__ __launch_bounds__(128) void k_lstm_recur(
    const _Float16* __restrict__ XG,    // [T][64][4096] f16 preactivations (incl bias)
    const _Float16* __restrict__ Wp,    // packed Whh fragments (K=1024, N=4096)
    float* __restrict__ cstate,         // [64][1024]
    _Float16* __restrict__ hbuf,        // [2][64][1024]
    _Float16* __restrict__ seq_out,     // store_all: [T][64][1024]; else [64][1024]
    int store_all,
    unsigned int* __restrict__ syncp)
{
    extern __shared__ _Float16 ldsW[];  // 65536 f16 = 128KB
    const int H = 1024;
    const int w = blockIdx.x;           // 0..63
    const int v = threadIdx.x >> 5;     // 0..3
    const int lane = threadIdx.x & 31;
    const int lo = lane >> 4;
    const int nl = lane & 15;
    const int m0 = v * 16;
    const int hc0 = w * 16;

    // ---- stage Whh slice into LDS (local frag l=g*32+kk <- global frag (g*64+w)*32+kk)
    {
        const uint4* src = (const uint4*)Wp;     // 64 x 16B chunks per fragment
        uint4* dst = (uint4*)ldsW;
        for (int c = (int)threadIdx.x; c < 8192; c += (int)blockDim.x) {
            const int l = c >> 6, rc = c & 63;
            const int g = l >> 5, kk = l & 31;
            dst[c] = src[(size_t)(((g * 64 + w) * 32 + kk) * 64 + rc)];
        }
    }
    __syncthreads();

    // per-gate LDS base (keeps ds imm offset = kk*1024B within 16 bits)
    const _Float16* lwg[4];
    #pragma unroll
    for (int g = 0; g < 4; ++g) lwg[g] = ldsW + g * 16384 + lane * 16;

    for (int t = 0; t < 512; ++t) {
        const _Float16* ap = hbuf + (size_t)(t & 1) * (64 * H)
                           + (size_t)(m0 + (lane & 15)) * H + 8 * lo;

        v8f acc[4];
        #pragma unroll
        for (int g = 0; g < 4; ++g) acc[g] = vzero8();

        #pragma unroll
        for (int kk = 0; kk < 32; ++kk) {
            FragA A;
            A.h[0] = *(const v8h*)(ap + kk * 32);
            A.h[1] = *(const v8h*)(ap + kk * 32 + 16);
            #pragma unroll
            for (int g = 0; g < 4; ++g) {
                FragA Bv;
                Bv.h[0] = *(const v8h*)(lwg[g] + kk * 512);
                Bv.h[1] = *(const v8h*)(lwg[g] + kk * 512 + 8);
                acc[g] = __builtin_amdgcn_wmma_f32_16x16x32_f16(
                    false, A.v, false, Bv.v, (short)0, acc[g], false, false);
            }
        }

        // LSTM cell update for this wave's 16x16 (batch x hcol) patch
        const _Float16* xgp = XG + ((size_t)t * 64 + m0 + 8 * lo) * 4096 + hc0 + nl;
        float* cp = cstate + (size_t)(m0 + 8 * lo) * H + hc0 + nl;
        _Float16* hnp = hbuf + (size_t)((t + 1) & 1) * (64 * H)
                      + (size_t)(m0 + 8 * lo) * H + hc0 + nl;
        _Float16* sp = seq_out
                     + (store_all ? (size_t)t * (64 * H) : (size_t)0)
                     + (size_t)(m0 + 8 * lo) * H + hc0 + nl;
        #pragma unroll
        for (int e = 0; e < 8; ++e) {
            const float gi = acc[0][e] + (float)xgp[e * 4096];
            const float gf = acc[1][e] + (float)xgp[e * 4096 + 1024];
            const float gg = acc[2][e] + (float)xgp[e * 4096 + 2048];
            const float go = acc[3][e] + (float)xgp[e * 4096 + 3072];
            const float ig = sigm(gi);
            const float fg = sigm(gf);
            const float gc = tanh_fast(gg);
            const float og = sigm(go);
            const float c = fg * cp[e * 1024] + ig * gc;
            cp[e * 1024] = c;
            const float h = og * tanh_fast(c);
            const _Float16 hh = (_Float16)h;
            hnp[e * 1024] = hh;
            if (store_all)     sp[e * 1024] = hh;
            else if (t == 511) sp[e * 1024] = hh;
        }

        // grid-wide barrier (monotonic counter)
        __syncthreads();
        if (threadIdx.x == 0) {
            __threadfence();
            atomicAdd(syncp, 1u);
            const unsigned target = 64u * (unsigned)(t + 1);
            while (__hip_atomic_load(syncp, __ATOMIC_ACQUIRE, __HIP_MEMORY_SCOPE_AGENT) < target)
                __builtin_amdgcn_s_sleep(1);
        }
        __syncthreads();
    }
}

// ---------------------------------------------------------------------------
// Final FC: out[b] = hT[b][:].Wfc + bfc   (one wave per batch row)
__global__ void k_fc(const _Float16* __restrict__ hT, const float* __restrict__ Wfc,
                     const float* __restrict__ bfc, float* __restrict__ out)
{
    const int b = blockIdx.x;
    const int lane = threadIdx.x;
    float s = 0.f;
    for (int n = lane; n < 1024; n += 32)
        s += (float)hT[(size_t)b * 1024 + n] * Wfc[n];
    #pragma unroll
    for (int off = 16; off > 0; off >>= 1)
        s += __shfl_down(s, off, 32);
    if (lane == 0) out[b] = s + bfc[0];
}

// ---------------------------------------------------------------------------
extern "C" void kernel_launch(void* const* d_in, const int* in_sizes, int n_in,
                              void* d_out, int out_size, void* d_ws, size_t ws_size,
                              hipStream_t stream) {
    (void)in_sizes; (void)n_in; (void)out_size;
    const int B = 64, T = 512, D = 256, H = 1024, G4 = 4096;

    const float* x    = (const float*)d_in[0];
    const float* Wih0 = (const float*)d_in[1];
    const float* Whh0 = (const float*)d_in[2];
    const float* b0   = (const float*)d_in[3];
    const float* Wih1 = (const float*)d_in[4];
    const float* Whh1 = (const float*)d_in[5];
    const float* b1   = (const float*)d_in[6];
    const float* Wfc  = (const float*)d_in[7];
    const float* bfc  = (const float*)d_in[8];
    float* out = (float*)d_out;

    char* ws = (char*)d_ws;
    size_t off = 0;
    auto alloc = [&](size_t bytes) { size_t o = off; off += (bytes + 255) & ~(size_t)255; return o; };
    const size_t XH   = alloc((size_t)B * T * D * 2);
    const size_t PW0  = alloc((size_t)G4 * D * 2);
    const size_t PWH0 = alloc((size_t)G4 * H * 2);
    const size_t PW1  = alloc((size_t)G4 * H * 2);
    const size_t PWH1 = alloc((size_t)G4 * H * 2);
    const size_t XG   = alloc((size_t)T * B * G4 * 2);
    const size_t SEQ0 = alloc((size_t)T * B * H * 2);
    const size_t HB0  = alloc((size_t)2 * B * H * 2);
    const size_t HB1  = alloc((size_t)2 * B * H * 2);
    const size_t C0   = alloc((size_t)B * H * 4);
    const size_t C1   = alloc((size_t)B * H * 4);
    const size_t HT   = alloc((size_t)B * H * 2);
    const size_t SYNC = alloc(512);
    if (off > ws_size) return;

    _Float16* xh    = (_Float16*)(ws + XH);
    _Float16* pw0   = (_Float16*)(ws + PW0);
    _Float16* pwh0  = (_Float16*)(ws + PWH0);
    _Float16* pw1   = (_Float16*)(ws + PW1);
    _Float16* pwh1  = (_Float16*)(ws + PWH1);
    _Float16* xg    = (_Float16*)(ws + XG);
    _Float16* seq0  = (_Float16*)(ws + SEQ0);
    _Float16* hb0   = (_Float16*)(ws + HB0);
    _Float16* hb1   = (_Float16*)(ws + HB1);
    float*    c0    = (float*)(ws + C0);
    float*    c1    = (float*)(ws + C1);
    _Float16* hT    = (_Float16*)(ws + HT);
    unsigned* sync0 = (unsigned*)(ws + SYNC);
    unsigned* sync1 = (unsigned*)(ws + SYNC + 256);

    {   // convert x to f16
        size_t n = (size_t)B * T * D;
        k_cvt_f16<<<dim3((unsigned)((n + 255) / 256)), dim3(256), 0, stream>>>(x, xh, n);
    }
    // pack weights into WMMA fragment order (f16)
    k_pack_w<<<dim3((G4 * D) / 256), dim3(256), 0, stream>>>(Wih0, pw0,  G4, D);
    k_pack_w<<<dim3((G4 * H) / 256), dim3(256), 0, stream>>>(Whh0, pwh0, G4, H);
    k_pack_w<<<dim3((G4 * H) / 256), dim3(256), 0, stream>>>(Wih1, pw1,  G4, H);
    k_pack_w<<<dim3((G4 * H) / 256), dim3(256), 0, stream>>>(Whh1, pwh1, G4, H);
    // zero states + sync counters
    k_zero_u32<<<dim3((2 * B * H / 2 + 255) / 256), dim3(256), 0, stream>>>((unsigned*)hb0, (size_t)2 * B * H / 2);
    k_zero_u32<<<dim3((2 * B * H / 2 + 255) / 256), dim3(256), 0, stream>>>((unsigned*)hb1, (size_t)2 * B * H / 2);
    k_zero_u32<<<dim3((B * H + 255) / 256), dim3(256), 0, stream>>>((unsigned*)c0, (size_t)B * H);
    k_zero_u32<<<dim3((B * H + 255) / 256), dim3(256), 0, stream>>>((unsigned*)c1, (size_t)B * H);
    k_zero_u32<<<dim3(1), dim3(128), 0, stream>>>((unsigned*)sync0, 128);

    // layer-0 input projection: xg = x @ Wih0^T + b0   (rows are [B,T])
    k_gemm_xg<<<dim3(4096), dim3(256), 0, stream>>>(xh, pw0, b0, xg, D, 1);
    // layer-0 recurrence (128KB dynamic LDS for Whh slice)
    k_lstm_recur<<<dim3(64), dim3(128), 131072, stream>>>(xg, pwh0, c0, hb0, seq0, 1, sync0);
    // layer-1 input projection: xg = seq0 @ Wih1^T + b1 (rows are [T,B])
    k_gemm_xg<<<dim3(4096), dim3(256), 0, stream>>>(seq0, pw1, b1, xg, H, 0);
    // layer-1 recurrence
    k_lstm_recur<<<dim3(64), dim3(128), 131072, stream>>>(xg, pwh1, c1, hb1, hT, 0, sync1);
    // FC head
    k_fc<<<dim3(64), dim3(32), 0, stream>>>(hT, Wfc, bfc, out);
}
